// RestormerFusion_4561255268513
// MI455X (gfx1250) — matrix operand order repaired and checked by reading, not compiled, not run on hardware
//
#include <hip/hip_runtime.h>
#include <hip/hip_bf16.h>
#include <math.h>

typedef _Float16 half_t;
typedef __attribute__((ext_vector_type(16))) _Float16 v16h;
typedef __attribute__((ext_vector_type(8)))  float    v8f;
typedef unsigned int uint32;

// pack two f32 -> two f16 in a dword
__device__ __forceinline__ uint32 pk2h(float a, float b) {
  union { half_t h[2]; uint32 u; } t;
  t.h[0] = (half_t)a; t.h[1] = (half_t)b;
  return t.u;
}

// ---------------------------------------------------------------------------
// WMMA fragment loaders (CDNA5 ISA 7.12.2 lane layouts, wave32)
// ---------------------------------------------------------------------------
// A fragment: 16x32 (MxK) f16.  m = lane%16.
//   lanes 0-15 : v0..3 -> K 0..7,  v4..7 -> K 16..23
//   lanes 16-31: v0..3 -> K 8..15, v4..7 -> K 24..31
__device__ __forceinline__ v16h frag_a_lds(const half_t* lds, int lane, int ld) {
  v16h f;
  uint32* fr = reinterpret_cast<uint32*>(&f);
  const int m  = lane & 15;
  const int kb = (lane & 16) ? 8 : 0;
  const uint32* row = reinterpret_cast<const uint32*>(lds + m * ld);
#pragma unroll
  for (int v = 0; v < 8; ++v) {
    int k = ((v & 4) ? 16 : 0) + kb + 2 * (v & 3);
    fr[v] = row[k >> 1];
  }
  return f;
}
// B fragment: 32x16 (KxN) f16, LDS tile stored N-major [n][k].
//   n = lane%16; lanes 0-15: K 0..7, 16..23; lanes 16-31: K 8..15, 24..31.
__device__ __forceinline__ v16h frag_b_lds(const half_t* lds, int lane, int ld) {
  v16h f;
  uint32* fr = reinterpret_cast<uint32*>(&f);
  const int n  = lane & 15;
  const int kb = (lane & 16) ? 8 : 0;
  const uint32* row = reinterpret_cast<const uint32*>(lds + n * ld);
#pragma unroll
  for (int v = 0; v < 8; ++v) fr[v] = row[(((v & 4) ? 16 : 0) + kb + 2 * (v & 3)) >> 1];
  return f;
}

__device__ __forceinline__ float gelu_f(float x) {
  return 0.5f * x * (1.0f + erff(x * 0.7071067811865476f));
}

// ---------------------------------------------------------------------------
// Generic batched WMMA GEMM:  C[M][N] = act( scale * A?T[M][K] * B[K][N] + bias ) + residual
// A: M x K row-major (transA=0) or K x M row-major (transA=1); staged to LDS f16.
// B: K x N row-major f32, staged to LDS f16 (N-major tile).
// Two-level batching: batch b -> (w = b/heads, h = b%heads) offsets.
// Block = 256 threads = 8 wave32 waves (2x4), tile M32 x N64, K-slab 32.
// ---------------------------------------------------------------------------
struct GemmArgs {
  const float* A; const float* B; float* C;
  const float* bias; const float* residual;
  int M, N, K;
  int lda, ldb, ldc;
  int transA;
  float scale;
  int act;                 // 0 = none, 1 = gelu
  int batches, heads;
  long long baseA, baseB, baseC;
  long long strideA_w, strideA_h;
  long long strideB_w, strideB_h;
  long long strideC_w, strideC_h;
};

#define LDA_S 40   // sA row stride (halves): 8B-aligned packed stores, conflict-free reads
#define LDB_S 36   // sB row stride (halves)

__global__ __launch_bounds__(256) void k_bgemm(GemmArgs g) {
  __shared__ __align__(16) half_t sA[32 * LDA_S];
  __shared__ __align__(16) half_t sB[64 * LDB_S];

  const int b  = blockIdx.z;
  const int wi = b / g.heads, hi = b % g.heads;
  const float* A = g.A + g.baseA + (long long)wi * g.strideA_w + (long long)hi * g.strideA_h;
  const float* B = g.B + g.baseB + (long long)wi * g.strideB_w + (long long)hi * g.strideB_h;
  float*       C = g.C + g.baseC + (long long)wi * g.strideC_w + (long long)hi * g.strideC_h;
  const float* R = g.residual ? g.residual + (long long)wi * g.strideC_w + (long long)hi * g.strideC_h
                              : (const float*)0;

  const int M = g.M, N = g.N, K = g.K;
  const int lda = g.lda, ldb = g.ldb;
  const int n0 = blockIdx.x * 64;
  const int m0 = blockIdx.y * 32;
  const int tid = threadIdx.x, lane = tid & 31, wave = tid >> 5;
  const int wm = wave >> 2, wn = wave & 3;                       // 2 x 4 wave grid

  const bool fullM = (m0 + 32 <= M);
  const bool fullN = (n0 + 64 <= N);

  v8f acc = {0.f, 0.f, 0.f, 0.f, 0.f, 0.f, 0.f, 0.f};

  for (int k0 = 0; k0 < K; k0 += 32) {
    const bool fullK = (k0 + 32 <= K);

    // ---------------- stage A (32 M x 32 K) ----------------
    if (!g.transA) {
      if (fullM && fullK) {
        // one b128 per thread: m = tid/8, k = (tid%8)*4
        const int m = tid >> 3, k = (tid & 7) * 4;
        const float4 v = *reinterpret_cast<const float4*>(&A[(long long)(m0 + m) * lda + (k0 + k)]);
        uint32* d = reinterpret_cast<uint32*>(&sA[m * LDA_S + k]);
        d[0] = pk2h(v.x, v.y);
        d[1] = pk2h(v.z, v.w);
      } else {
        float va[4]; int pm[4], pk[4];
#pragma unroll
        for (int u = 0; u < 4; ++u) {
          int i = tid + u * 256;
          pm[u] = i >> 5; pk[u] = i & 31;
          int mm = (m0 + pm[u] < M) ? (m0 + pm[u]) : (M - 1);
          int kk = (k0 + pk[u] < K) ? (k0 + pk[u]) : (K - 1);
          va[u] = A[(long long)mm * lda + kk];           // clamped, branchless
        }
#pragma unroll
        for (int u = 0; u < 4; ++u) {
          bool inb = (m0 + pm[u] < M) && (k0 + pk[u] < K);
          sA[pm[u] * LDA_S + pk[u]] = (half_t)(inb ? va[u] : 0.f);
        }
      }
    } else { // A stored K x M
      if (fullM && fullK) {
        const int k = tid >> 3, m = (tid & 7) * 4;       // b128 along m
        const float4 v = *reinterpret_cast<const float4*>(&A[(long long)(k0 + k) * lda + (m0 + m)]);
        sA[(m + 0) * LDA_S + k] = (half_t)v.x;
        sA[(m + 1) * LDA_S + k] = (half_t)v.y;
        sA[(m + 2) * LDA_S + k] = (half_t)v.z;
        sA[(m + 3) * LDA_S + k] = (half_t)v.w;
      } else {
        float va[4]; int pm[4], pk[4];
#pragma unroll
        for (int u = 0; u < 4; ++u) {
          int i = tid + u * 256;
          pm[u] = i & 31; pk[u] = i >> 5;
          int mm = (m0 + pm[u] < M) ? (m0 + pm[u]) : (M - 1);
          int kk = (k0 + pk[u] < K) ? (k0 + pk[u]) : (K - 1);
          va[u] = A[(long long)kk * lda + mm];
        }
#pragma unroll
        for (int u = 0; u < 4; ++u) {
          bool inb = (m0 + pm[u] < M) && (k0 + pk[u] < K);
          sA[pm[u] * LDA_S + pk[u]] = (half_t)(inb ? va[u] : 0.f);
        }
      }
    }

    // ---------------- stage B (32 K x 64 N), write N-major ----------------
    if (fullN && fullK) {
      const int k = tid >> 3, n = (tid & 7) * 8;         // two b128 along n
      const float* src = &B[(long long)(k0 + k) * ldb + (n0 + n)];
      const float4 v0 = *reinterpret_cast<const float4*>(src);
      const float4 v1 = *reinterpret_cast<const float4*>(src + 4);
      sB[(n + 0) * LDB_S + k] = (half_t)v0.x;
      sB[(n + 1) * LDB_S + k] = (half_t)v0.y;
      sB[(n + 2) * LDB_S + k] = (half_t)v0.z;
      sB[(n + 3) * LDB_S + k] = (half_t)v0.w;
      sB[(n + 4) * LDB_S + k] = (half_t)v1.x;
      sB[(n + 5) * LDB_S + k] = (half_t)v1.y;
      sB[(n + 6) * LDB_S + k] = (half_t)v1.z;
      sB[(n + 7) * LDB_S + k] = (half_t)v1.w;
    } else {
      float vb[8]; int pk[8], pn[8];
#pragma unroll
      for (int u = 0; u < 8; ++u) {
        int i = tid + u * 256;
        pk[u] = i >> 6; pn[u] = i & 63;
        int kk = (k0 + pk[u] < K) ? (k0 + pk[u]) : (K - 1);
        int nn = (n0 + pn[u] < N) ? (n0 + pn[u]) : (N - 1);
        vb[u] = B[(long long)kk * ldb + nn];
      }
#pragma unroll
      for (int u = 0; u < 8; ++u) {
        bool inb = (k0 + pk[u] < K) && (n0 + pn[u] < N);
        sB[pn[u] * LDB_S + pk[u]] = (half_t)(inb ? vb[u] : 0.f);
      }
    }

    // L2 prefetch hint for the next K slab of B
    if (k0 + 32 < K && tid < 64 && n0 + tid < N)
      __builtin_prefetch(&B[(long long)(k0 + 32) * ldb + (n0 + tid)], 0, 1);
    __syncthreads();

    v16h af = frag_a_lds(sA + wm * 16 * LDA_S, lane, LDA_S);
    v16h bf = frag_b_lds(sB + wn * 16 * LDB_S, lane, LDB_S);
    acc = __builtin_amdgcn_wmma_f32_16x16x32_f16(false, af, false, bf,
                                                 (short)0, acc, false, false);
    __syncthreads();
  }

  // ---- epilogue: C layout (ISA): vgpr r -> row r + (lane>=16 ? 8:0), col = lane%16 ----
  const int col   = n0 + wn * 16 + (lane & 15);
  const int rbase = m0 + wm * 16 + ((lane & 16) ? 8 : 0);
  if (col < N) {
#pragma unroll
    for (int r = 0; r < 8; ++r) {
      int row = rbase + r;
      if (row < M) {
        float v = acc[r] * g.scale;
        if (g.bias) v += g.bias[row];
        if (g.act == 1) v = gelu_f(v);
        long long off = (long long)row * g.ldc + col;
        if (R) v += R[off];
        C[off] = v;
      }
    }
  }
}

// ---------------------------------------------------------------------------
// Conditional LayerNorm (over channels) + FiLM.  film = [gamma(C), beta(C)].
// ---------------------------------------------------------------------------
__global__ void k_cond_ln(const float* __restrict__ x, float* __restrict__ y,
                          const float* __restrict__ lw, const float* __restrict__ lb,
                          const float* __restrict__ film, int C, int P) {
  int p = blockIdx.x * blockDim.x + threadIdx.x;
  if (p >= P) return;
  float s = 0.f, s2 = 0.f;
  for (int c = 0; c < C; ++c) { float v = x[(long long)c * P + p]; s += v; s2 += v * v; }
  float mu = s / C;
  float var = s2 / C - mu * mu;
  float rs = rsqrtf(var + 1e-6f);
  for (int c = 0; c < C; ++c) {
    float v = (x[(long long)c * P + p] - mu) * rs * lw[c] + lb[c];
    y[(long long)c * P + p] = film[c] * v + film[C + c];
  }
}

// Column softmax over rows (t axis) of Z[batch][Mr][Ncols]; optional window rel-bias.
__global__ void k_softmax_col(float* __restrict__ Z, int Mr, int Ncols, int heads,
                              const float* __restrict__ rel_bias) {
  long long b = blockIdx.y;
  int h = (int)(b % heads);
  float* z = Z + b * (long long)Mr * Ncols;
  int s = blockIdx.x * blockDim.x + threadIdx.x;
  if (s >= Ncols) return;
  if (rel_bias) { // window attention: Mr == Ncols == 64, Z[t][s] = attn[s][t]
    int si = s >> 3, sj = s & 7;
    for (int t = 0; t < Mr; ++t) {
      int ti = t >> 3, tj = t & 7;
      int idx = (si - ti + 7) * 15 + (sj - tj + 7);
      z[(long long)t * Ncols + s] += rel_bias[idx * heads + h];
    }
  }
  float mx = -3.4e38f;
  for (int t = 0; t < Mr; ++t) mx = fmaxf(mx, z[(long long)t * Ncols + s]);
  float sum = 0.f;
  for (int t = 0; t < Mr; ++t) {
    float e = __expf(z[(long long)t * Ncols + s] - mx);
    z[(long long)t * Ncols + s] = e; sum += e;
  }
  float inv = 1.f / sum;
  for (int t = 0; t < Mr; ++t) z[(long long)t * Ncols + s] *= inv;
}

// Window gather/scatter (8x8 windows, optional cyclic shift).
__global__ void k_win_gather(const float* __restrict__ img, float* __restrict__ out,
                             int C, int H, int W, int shift) {
  int total = C * H * W;
  int i = blockIdx.x * blockDim.x + threadIdx.x;
  if (i >= total) return;
  int WW = W >> 3;
  int s = i & 63, t = i >> 6;
  int c = t % C, win = t / C;
  int wy = win / WW, wx = win % WW;
  int r  = (wy * 8 + (s >> 3) + shift) & (H - 1);
  int cc = (wx * 8 + (s & 7)  + shift) & (W - 1);
  out[i] = img[(long long)c * H * W + r * W + cc];
}
__global__ void k_win_scatter(const float* __restrict__ win_buf, float* __restrict__ img,
                              int C, int H, int W, int shift) {
  int total = C * H * W;
  int i = blockIdx.x * blockDim.x + threadIdx.x;
  if (i >= total) return;
  int WW = W >> 3;
  int s = i & 63, t = i >> 6;
  int c = t % C, win = t / C;
  int wy = win / WW, wx = win % WW;
  int r  = (wy * 8 + (s >> 3) + shift) & (H - 1);
  int cc = (wx * 8 + (s & 7)  + shift) & (W - 1);
  img[(long long)c * H * W + r * W + cc] = win_buf[i];
}

// Direct 3x3 conv, SAME zero pad (shallow stem + output head; tiny FLOP share).
__global__ void k_conv3x3(const float* __restrict__ x, const float* __restrict__ w,
                          float* __restrict__ y, int Cin, int Cout, int H, int W) {
  int P = H * W;
  int i = blockIdx.x * blockDim.x + threadIdx.x;
  if (i >= Cout * P) return;
  int o = i / P, p = i % P, r = p / W, c = p % W;
  float acc = 0.f;
  for (int ci = 0; ci < Cin; ++ci)
    for (int dy = 0; dy < 3; ++dy) {
      int rr = r + dy - 1; if (rr < 0 || rr >= H) continue;
      for (int dx = 0; dx < 3; ++dx) {
        int cc = c + dx - 1; if (cc < 0 || cc >= W) continue;
        acc += w[((o * Cin + ci) * 3 + dy) * 3 + dx] * x[(long long)ci * P + rr * W + cc];
      }
    }
  y[i] = acc;
}

__global__ void k_unshuffle(const float* __restrict__ x, float* __restrict__ y,
                            int C, int H, int W) { // y[4C][H/2][W/2]
  int H2 = H >> 1, W2 = W >> 1;
  int total = 4 * C * H2 * W2;
  int i = blockIdx.x * blockDim.x + threadIdx.x;
  if (i >= total) return;
  int p = i % (H2 * W2), oc = i / (H2 * W2);
  int s2 = oc & 1, s1 = (oc >> 1) & 1, c = oc >> 2;
  int h2 = p / W2, w2 = p % W2;
  y[i] = x[(long long)c * H * W + (2 * h2 + s1) * W + (2 * w2 + s2)];
}
__global__ void k_shuffle(const float* __restrict__ x, float* __restrict__ y,
                          int C, int H, int W) { // y[C/4][2H][2W]
  int HP = 2 * H, WP = 2 * W;
  int total = C * H * W;
  int i = blockIdx.x * blockDim.x + threadIdx.x;
  if (i >= total) return;
  int p = i % (HP * WP), c4 = i / (HP * WP);
  int r = p / WP, cc = p % WP;
  int s1 = r & 1, s2 = cc & 1;
  y[i] = x[(long long)(c4 * 4 + s1 * 2 + s2) * H * W + (r >> 1) * W + (cc >> 1)];
}

// Meta FiLM MLP: h = relu(meta @ w1.T + b1), film = h @ w2.T + b2  -> [gamma(D), beta(D)]
__global__ void k_meta(const float* __restrict__ meta,
                       const float* __restrict__ w1, const float* __restrict__ b1,
                       const float* __restrict__ w2, const float* __restrict__ b2,
                       float* __restrict__ film, int D) {
  __shared__ float h[384];
  int t = threadIdx.x;
  if (t < D) {
    float a = b1[t];
    for (int i = 0; i < 3; ++i) a += meta[i] * w1[t * 3 + i];
    h[t] = fmaxf(a, 0.f);
  }
  __syncthreads();
  if (t < 2 * D) {
    float a = b2[t];
    for (int j = 0; j < D; ++j) a += h[j] * w2[t * D + j];
    film[t] = a;
  }
}

// ---------------------------------------------------------------------------
// Host-side orchestration
// ---------------------------------------------------------------------------
struct MetaP  { const float *w1, *b1, *w2, *b2; };
struct Fusion { const float *n1w,*n1b,*n2w,*n2b,*qkv_w,*qkv_b,*proj_w,*proj_b,*rel,
                            *cq_w,*ckv_w,*cproj_w,*cproj_b,*ffn1,*ffn2; };
struct GBlock { const float *n1w,*n1b,*n2w,*n2b,*qkv_w,*proj_w,*proj_b,*ffn1,*ffn2; };

struct Ctx {
  hipStream_t st;
  float *XR, *XS, *SK[3];
  float *T_LN, *T_X1, *T_X2, *T_QKV, *T_GATH, *T_Z, *T_O, *T_CAT, *T_PS, *T_FILM, *T_FFN;
};

static inline void launch_bgemm(const Ctx& c, const GemmArgs& g) {
  dim3 grid((unsigned)((g.N + 63) / 64), (unsigned)((g.M + 31) / 32), (unsigned)g.batches);
  k_bgemm<<<grid, dim3(256), 0, c.st>>>(g);
}
static inline GemmArgs G0(const float* A, const float* B, float* C, int M, int N, int K) {
  GemmArgs g{}; g.A = A; g.B = B; g.C = C; g.M = M; g.N = N; g.K = K;
  g.lda = K; g.ldb = N; g.ldc = N; g.scale = 1.f; g.batches = 1; g.heads = 1;
  return g;
}
static inline void run_ln(const Ctx& c, const float* x, float* y,
                          const float* w, const float* b, int C, int P) {
  k_cond_ln<<<dim3((P + 255) / 256), dim3(256), 0, c.st>>>(x, y, w, b, c.T_FILM, C, P);
}
static inline void run_gather(const Ctx& c, const float* img, float* out,
                              int Cc, int H, int W, int shift) {
  int total = Cc * H * W;
  k_win_gather<<<dim3((total + 255) / 256), dim3(256), 0, c.st>>>(img, out, Cc, H, W, shift);
}
static inline void run_scatter(const Ctx& c, const float* buf, float* img,
                               int Cc, int H, int W, int shift) {
  int total = Cc * H * W;
  k_win_scatter<<<dim3((total + 255) / 256), dim3(256), 0, c.st>>>(buf, img, Cc, H, W, shift);
}

static void fusion_fwd(const Ctx& c, float* dst, const float* xin, const float* xkv,
                       const Fusion& f, int C, int heads, int H, int W, int shift) {
  const int P = H * W;
  const int d = C / heads;
  const int nwin = (H / 8) * (W / 8);
  const float scl = 1.0f / sqrtf((float)d);

  // ---------------- window attention ----------------
  run_ln(c, xin, c.T_LN, f.n1w, f.n1b, C, P);
  { GemmArgs g = G0(f.qkv_w, c.T_LN, c.T_QKV, 3 * C, P, C); g.bias = f.qkv_b; launch_bgemm(c, g); }
  run_gather(c, c.T_QKV, c.T_GATH, 3 * C, H, W, shift);
  { // Z = K^T * Q (scaled), batched over (win, head)
    GemmArgs g{};
    g.A = c.T_GATH; g.baseA = (long long)C * 64; g.transA = 1; g.lda = 64;
    g.strideA_w = (long long)3 * C * 64; g.strideA_h = (long long)d * 64;
    g.B = c.T_GATH; g.ldb = 64;
    g.strideB_w = (long long)3 * C * 64; g.strideB_h = (long long)d * 64;
    g.C = c.T_Z; g.ldc = 64;
    g.strideC_w = (long long)heads * 64 * 64; g.strideC_h = 64 * 64;
    g.M = 64; g.N = 64; g.K = d; g.scale = scl;
    g.batches = nwin * heads; g.heads = heads;
    launch_bgemm(c, g);
  }
  k_softmax_col<<<dim3(1, nwin * heads), dim3(64), 0, c.st>>>(c.T_Z, 64, 64, heads, f.rel);
  { // O = V * Z
    GemmArgs g{};
    g.A = c.T_GATH; g.baseA = (long long)2 * C * 64; g.lda = 64;
    g.strideA_w = (long long)3 * C * 64; g.strideA_h = (long long)d * 64;
    g.B = c.T_Z; g.ldb = 64;
    g.strideB_w = (long long)heads * 64 * 64; g.strideB_h = 64 * 64;
    g.C = c.T_O; g.ldc = 64;
    g.strideC_w = (long long)C * 64; g.strideC_h = (long long)d * 64;
    g.M = d; g.N = 64; g.K = 64; g.scale = 1.f;
    g.batches = nwin * heads; g.heads = heads;
    launch_bgemm(c, g);
  }
  run_scatter(c, c.T_O, c.T_X2, C, H, W, shift);
  { GemmArgs g = G0(f.proj_w, c.T_X2, c.T_X1, C, P, C);
    g.bias = f.proj_b; g.residual = xin; launch_bgemm(c, g); }   // x1

  // ---------------- cross attention ----------------
  run_ln(c, c.T_X1, c.T_LN, f.n1w, f.n1b, C, P);
  { GemmArgs g = G0(f.cq_w,  c.T_LN, c.T_QKV, C, P, C); launch_bgemm(c, g); }
  { GemmArgs g = G0(f.ckv_w, xkv, c.T_QKV + (size_t)C * P, 2 * C, P, C); launch_bgemm(c, g); }
  run_gather(c, c.T_QKV, c.T_GATH, C, H, W, 0);
  run_gather(c, c.T_QKV + (size_t)C * P, c.T_GATH + (size_t)C * P, 2 * C, H, W, 0);
  float* GQ  = c.T_GATH;
  float* GKV = c.T_GATH + (size_t)C * P;
  { GemmArgs g{};
    g.A = GKV; g.transA = 1; g.lda = 64;
    g.strideA_w = (long long)2 * C * 64; g.strideA_h = (long long)d * 64;
    g.B = GQ; g.ldb = 64;
    g.strideB_w = (long long)C * 64; g.strideB_h = (long long)d * 64;
    g.C = c.T_Z; g.ldc = 64;
    g.strideC_w = (long long)heads * 64 * 64; g.strideC_h = 64 * 64;
    g.M = 64; g.N = 64; g.K = d; g.scale = scl;
    g.batches = nwin * heads; g.heads = heads;
    launch_bgemm(c, g);
  }
  k_softmax_col<<<dim3(1, nwin * heads), dim3(64), 0, c.st>>>(c.T_Z, 64, 64, heads,
                                                             (const float*)nullptr);
  { GemmArgs g{};
    g.A = GKV; g.baseA = (long long)C * 64; g.lda = 64;
    g.strideA_w = (long long)2 * C * 64; g.strideA_h = (long long)d * 64;
    g.B = c.T_Z; g.ldb = 64;
    g.strideB_w = (long long)heads * 64 * 64; g.strideB_h = 64 * 64;
    g.C = c.T_O; g.ldc = 64;
    g.strideC_w = (long long)C * 64; g.strideC_h = (long long)d * 64;
    g.M = d; g.N = 64; g.K = 64; g.scale = 1.f;
    g.batches = nwin * heads; g.heads = heads;
    launch_bgemm(c, g);
  }
  run_scatter(c, c.T_O, c.T_LN, C, H, W, 0);
  { GemmArgs g = G0(f.cproj_w, c.T_LN, c.T_X2, C, P, C);
    g.bias = f.cproj_b; g.residual = c.T_X1; launch_bgemm(c, g); } // x2

  // ---------------- FFN ----------------
  run_ln(c, c.T_X2, c.T_LN, f.n2w, f.n2b, C, P);
  { GemmArgs g = G0(f.ffn1, c.T_LN, c.T_FFN, 4 * C, P, C); g.act = 1; launch_bgemm(c, g); }
  { GemmArgs g = G0(f.ffn2, c.T_FFN, dst, C, P, 4 * C); g.residual = c.T_X2; launch_bgemm(c, g); }
}

static void gblock_fwd(const Ctx& c, float* dst, const float* xin, const GBlock& gb,
                       int C, int heads, int n) {
  const int d = C / heads;
  const float scl = 1.0f / sqrtf((float)d);
  run_ln(c, xin, c.T_LN, gb.n1w, gb.n1b, C, n);
  { GemmArgs g = G0(gb.qkv_w, c.T_LN, c.T_QKV, 3 * C, n, C); launch_bgemm(c, g); }
  { GemmArgs g{};
    g.A = c.T_QKV; g.baseA = (long long)C * n; g.transA = 1; g.lda = n;
    g.strideA_h = (long long)d * n;
    g.B = c.T_QKV; g.ldb = n; g.strideB_h = (long long)d * n;
    g.C = c.T_Z; g.ldc = n; g.strideC_h = (long long)n * n;
    g.M = n; g.N = n; g.K = d; g.scale = scl;
    g.batches = heads; g.heads = heads;
    launch_bgemm(c, g);
  }
  k_softmax_col<<<dim3((n + 127) / 128, heads), dim3(128), 0, c.st>>>(c.T_Z, n, n, heads,
                                                                     (const float*)nullptr);
  { GemmArgs g{};
    g.A = c.T_QKV; g.baseA = (long long)2 * C * n; g.lda = n; g.strideA_h = (long long)d * n;
    g.B = c.T_Z; g.ldb = n; g.strideB_h = (long long)n * n;
    g.C = c.T_O; g.ldc = n; g.strideC_h = (long long)d * n;
    g.M = d; g.N = n; g.K = n; g.scale = 1.f;
    g.batches = heads; g.heads = heads;
    launch_bgemm(c, g);
  }
  { GemmArgs g = G0(gb.proj_w, c.T_O, c.T_X1, C, n, C);
    g.bias = gb.proj_b; g.residual = xin; launch_bgemm(c, g); }
  run_ln(c, c.T_X1, c.T_LN, gb.n2w, gb.n2b, C, n);
  { GemmArgs g = G0(gb.ffn1, c.T_LN, c.T_FFN, 4 * C, n, C); g.act = 1; launch_bgemm(c, g); }
  { GemmArgs g = G0(gb.ffn2, c.T_FFN, dst, C, n, 4 * C); g.residual = c.T_X1; launch_bgemm(c, g); }
}

extern "C" void kernel_launch(void* const* d_in, const int* in_sizes, int n_in,
                              void* d_out, int out_size, void* d_ws, size_t ws_size,
                              hipStream_t stream) {
  (void)in_sizes; (void)n_in; (void)out_size; (void)ws_size;

  // ---- parse flattened inputs (setup_inputs dict order, depth-first) ----
  int ci = 0;
  auto nx = [&]() -> const float* { return (const float*)d_in[ci++]; };
  const float* rgb  = nx();
  const float* sar  = nx();
  const float* meta = nx();
  const float* sh_rgb = nx();
  const float* sh_sar = nx();
  MetaP mp[4];
  for (int i = 0; i < 4; ++i) { mp[i].w1 = nx(); mp[i].b1 = nx(); mp[i].w2 = nx(); mp[i].b2 = nx(); }
  auto read_fusion = [&](Fusion& f) {
    f.n1w = nx(); f.n1b = nx(); f.n2w = nx(); f.n2b = nx();
    f.qkv_w = nx(); f.qkv_b = nx(); f.proj_w = nx(); f.proj_b = nx();
    f.rel = nx(); f.cq_w = nx(); f.ckv_w = nx(); f.cproj_w = nx(); f.cproj_b = nx();
    f.ffn1 = nx(); f.ffn2 = nx();
  };
  Fusion enc[4][2];
  for (int i = 0; i < 4; ++i) for (int b = 0; b < 2; ++b) read_fusion(enc[i][b]);
  const float* downs[3]; for (int i = 0; i < 3; ++i) downs[i] = nx();
  GBlock bot[2];
  for (int i = 0; i < 2; ++i) {
    GBlock& g = bot[i];
    g.n1w = nx(); g.n1b = nx(); g.n2w = nx(); g.n2b = nx();
    g.qkv_w = nx(); g.proj_w = nx(); g.proj_b = nx(); g.ffn1 = nx(); g.ffn2 = nx();
  }
  const float* ups[3]; for (int i = 0; i < 3; ++i) ups[i] = nx();
  Fusion dec[3][2];
  for (int j = 0; j < 3; ++j) for (int b = 0; b < 2; ++b) read_fusion(dec[j][b]);
  const float* outw = nx();

  // ---- carve workspace (bump allocator; all activations live in 192MB-L2 scale) ----
  char* wsb = (char*)d_ws;
  size_t cur = 0;
  auto alloc = [&](size_t elems) -> float* {
    float* p = (float*)(wsb + cur);
    cur += ((elems * sizeof(float)) + 255) & ~(size_t)255;
    return p;
  };
  const size_t E = (size_t)48 * 256 * 256;  // 3,145,728 (max C*P, constant per level pair)
  Ctx c{};
  c.st     = stream;
  c.XR     = alloc(E);
  c.XS     = alloc(E);
  c.SK[0]  = alloc(E);
  c.SK[1]  = alloc(E / 2);
  c.SK[2]  = alloc(E / 4);
  c.T_LN   = alloc(E);
  c.T_X1   = alloc(E);
  c.T_X2   = alloc(E);
  c.T_QKV  = alloc(3 * E);          // contiguous with T_GATH (3E is 256B aligned)
  c.T_GATH = alloc(3 * E);
  c.T_Z    = alloc((size_t)8 * 1024 * 1024);
  c.T_O    = alloc(E);
  c.T_CAT  = alloc(E + E / 2);
  c.T_PS   = alloc(E);
  c.T_FILM = alloc(1024);
  c.T_FFN  = c.T_QKV;               // FFN hidden (4E) overlays QKV+GATH region (6E)

  const int DIMSv[4]  = {48, 96, 192, 384};
  const int HEADSv[4] = {1, 2, 4, 8};

  // ---- shallow stems ----
  k_conv3x3<<<dim3((48 * 256 * 256 + 255) / 256), dim3(256), 0, stream>>>(rgb, sh_rgb, c.XR, 3, 48, 256, 256);
  k_conv3x3<<<dim3((48 * 256 * 256 + 255) / 256), dim3(256), 0, stream>>>(sar, sh_sar, c.XS, 3, 48, 256, 256);

  // ---- encoder ----
  for (int i = 0; i < 4; ++i) {
    int C = DIMSv[i], H = 256 >> i;
    k_meta<<<dim3(1), dim3(2 * C), 0, stream>>>(meta, mp[i].w1, mp[i].b1, mp[i].w2, mp[i].b2,
                                                c.T_FILM, C);
    int shift = (i & 1) ? 4 : 0;
    for (int b = 0; b < 2; ++b) {
      fusion_fwd(c, c.XR, c.XR, c.XS, enc[i][b], C, HEADSv[i], H, H, shift);
      fusion_fwd(c, c.XS, c.XS, c.XS, enc[i][b], C, HEADSv[i], H, H, shift);
    }
    if (i < 3) {
      size_t bytes = (size_t)C * H * H * sizeof(float);
      hipMemcpyAsync(c.SK[i], c.XS, bytes, hipMemcpyDeviceToDevice, stream);
      int P2 = (H / 2) * (H / 2), total = 4 * C * P2;
      k_unshuffle<<<dim3((total + 255) / 256), dim3(256), 0, stream>>>(c.XR, c.T_PS, C, H, H);
      { GemmArgs g = G0(downs[i], c.T_PS, c.XR, DIMSv[i + 1], P2, 4 * C); launch_bgemm(c, g); }
      k_unshuffle<<<dim3((total + 255) / 256), dim3(256), 0, stream>>>(c.XS, c.T_PS, C, H, H);
      { GemmArgs g = G0(downs[i], c.T_PS, c.XS, DIMSv[i + 1], P2, 4 * C); launch_bgemm(c, g); }
    }
  }

  // ---- bottleneck (global MDTA) ----
  k_meta<<<dim3(1), dim3(2 * 384), 0, stream>>>(meta, mp[3].w1, mp[3].b1, mp[3].w2, mp[3].b2,
                                                c.T_FILM, 384);
  for (int b = 0; b < 2; ++b) gblock_fwd(c, c.XR, c.XR, bot[b], 384, 8, 32 * 32);

  // ---- decoder ----
  for (int j = 0; j < 3; ++j) {
    int level = 2 - j;
    int C = DIMSv[level], Cup = DIMSv[level + 1];
    int H = 256 >> level, P = H * H;
    k_meta<<<dim3(1), dim3(2 * C), 0, stream>>>(meta, mp[level].w1, mp[level].b1,
                                                mp[level].w2, mp[level].b2, c.T_FILM, C);
    // xr = conv1x1(concat([pixel_shuffle(xr), skip]), ups[j])
    int totalS = Cup * (H / 2) * (H / 2);
    k_shuffle<<<dim3((totalS + 255) / 256), dim3(256), 0, stream>>>(c.XR, c.T_CAT, Cup, H / 2, H / 2);
    hipMemcpyAsync(c.T_CAT + (size_t)(Cup / 4) * P, c.SK[level],
                   (size_t)C * P * sizeof(float), hipMemcpyDeviceToDevice, stream);
    { GemmArgs g = G0(ups[j], c.T_CAT, c.XR, C, P, Cup / 4 + C); launch_bgemm(c, g); }
    int shift = (level & 1) ? 4 : 0;
    for (int b = 0; b < 2; ++b)
      fusion_fwd(c, c.XR, c.XR, c.SK[level], dec[j][b], C, HEADSv[level], H, H, shift);
  }

  // ---- output head ----
  k_conv3x3<<<dim3((3 * 256 * 256 + 255) / 256), dim3(256), 0, stream>>>(c.XR, outw,
                                                                         (float*)d_out, 48, 3, 256, 256);
}
